// CrossAttentionBlock_70514773066111
// MI455X (gfx1250) — matrix operand order repaired and checked by reading, not compiled, not run on hardware
//
#include <hip/hip_runtime.h>

// ---------------------------------------------------------------------------
// CDNA5 (gfx1250) cross-attention block.
// All GEMM-like math runs through v_wmma_f32_16x16x32_f16 (wave32).
// GEMM: double-buffered software pipeline; A-tiles staged with
// global_load_async_to_lds_b128 (ASYNCcnt path), B-tiles prefetched into
// registers during compute.
// ---------------------------------------------------------------------------

typedef __attribute__((ext_vector_type(16))) _Float16 v16h;
typedef __attribute__((ext_vector_type(8)))  float    v8f;

__device__ inline v8f v8f_zero() {
  v8f z;
#pragma unroll
  for (int i = 0; i < 8; ++i) z[i] = 0.0f;
  return z;
}

__device__ inline v8f wmma_f16(v16h a, v16h b, v8f c) {
  // D = A(16x32 f16) * B(32x16 f16) + C(16x16 f32)
  return __builtin_amdgcn_wmma_f32_16x16x32_f16(
      /*neg_a=*/false, a, /*neg_b=*/false, b,
      /*c_mod=*/(short)0, c, /*reuse_a=*/false, /*reuse_b=*/false);
}

// Async copy of 32 contiguous bytes from global to LDS (per lane).
// INST_OFFSET is added to BOTH the LDS and the global address (ISA ch10.4).
__device__ inline void async_copy_b256(const _Float16* gsrc, _Float16* lds_dst) {
  unsigned lds_addr = (unsigned)(unsigned long long)lds_dst;  // LDS = addr[31:0]
  asm volatile(
      "global_load_async_to_lds_b128 %0, %1, off\n\t"
      "global_load_async_to_lds_b128 %0, %1, off offset:16"
      :
      : "v"(lds_addr), "v"(gsrc)
      : "memory");
}

__device__ inline void wait_async_all() {
  asm volatile("s_wait_asynccnt 0x0" ::: "memory");
}

// ---------------------------------------------------------------------------
// LayerNorm over last dim (768) -> f16. One 256-thread block per row.
// ---------------------------------------------------------------------------
__global__ void __launch_bounds__(256) ln_f16_kernel(
    const float* __restrict__ x, const float* __restrict__ gamma,
    const float* __restrict__ beta, _Float16* __restrict__ y) {
  __shared__ float red[256];
  const size_t row = blockIdx.x;
  const float* xr = x + row * 768;
  const int t = threadIdx.x;

  float v0 = xr[t], v1 = xr[t + 256], v2 = xr[t + 512];
  red[t] = v0 + v1 + v2;
  __syncthreads();
#pragma unroll
  for (int off = 128; off > 0; off >>= 1) {
    if (t < off) red[t] += red[t + off];
    __syncthreads();
  }
  const float mu = red[0] * (1.0f / 768.0f);
  __syncthreads();

  const float d0 = v0 - mu, d1 = v1 - mu, d2 = v2 - mu;
  red[t] = d0 * d0 + d1 * d1 + d2 * d2;
  __syncthreads();
#pragma unroll
  for (int off = 128; off > 0; off >>= 1) {
    if (t < off) red[t] += red[t + off];
    __syncthreads();
  }
  const float var = red[0] * (1.0f / 768.0f);
  const float rstd = rsqrtf(var + 1e-5f);

  _Float16* yr = y + row * 768;
  yr[t]       = (_Float16)(d0 * rstd * gamma[t]       + beta[t]);
  yr[t + 256] = (_Float16)(d1 * rstd * gamma[t + 256] + beta[t + 256]);
  yr[t + 512] = (_Float16)(d2 * rstd * gamma[t + 512] + beta[t + 512]);
}

// ---------------------------------------------------------------------------
// f32 -> f16 elementwise (weight conversion)
// ---------------------------------------------------------------------------
__global__ void f32_to_f16_kernel(const float* __restrict__ x,
                                  _Float16* __restrict__ y, int n) {
  int i = blockIdx.x * blockDim.x + threadIdx.x;
  if (i < n) y[i] = (_Float16)x[i];
}

// ---------------------------------------------------------------------------
// WMMA GEMM:  C[M,N] = A[M,K](f16,row-major) * B[K,N](f16,row-major) + bias[N]
// Block tile 128x128, K step 32, double-buffered pipeline. 8 waves in
// 4(M)x2(N); each wave owns a 32x64 macro tile = 2x4 WMMA tiles.
// A staged async global->LDS (no VGPR round-trip); B prefetched to registers
// during compute, then scattered transposed ([n][k]) so the WMMA B-fragment
// (column per lane, K contiguous) is a contiguous LDS read.
// ---------------------------------------------------------------------------
template <bool OUT_F32>
__global__ void __launch_bounds__(256) gemm_bias_kernel(
    const _Float16* __restrict__ A, int lda,
    const _Float16* __restrict__ B, int ldb,
    const float* __restrict__ bias,
    float* __restrict__ Cf, _Float16* __restrict__ Ch, int ldc, int K) {
  __shared__ _Float16 As[2][128][40];  // 128x32 tiles, padded
  __shared__ _Float16 Bs[2][128][40];  // transposed 32x128 tiles -> [n][k]

  const int lane = threadIdx.x & 31;
  const int w    = threadIdx.x >> 5;
  const int wm   = (w & 3) * 32;   // 4 waves along M
  const int wn   = (w >> 2) * 64;  // 2 waves along N
  const int m0   = blockIdx.y * 128;
  const int n0   = blockIdx.x * 128;

  const int arow  = lane & 15;
  const int akoff = (lane >> 4) * 16;   // A frag: K 0..15 / 16..31
  const int bcol  = lane & 15;
  const int bkoff = (lane >> 4) * 16;   // B frag: K 0..15 / 16..31

  v8f acc[2][4];
#pragma unroll
  for (int mi = 0; mi < 2; ++mi)
#pragma unroll
    for (int ni = 0; ni < 4; ++ni) acc[mi][ni] = v8f_zero();

  // per-thread staging coordinates
  const int a_row = threadIdx.x >> 1, a_seg = threadIdx.x & 1;
  const int b_krow = threadIdx.x >> 3, b_seg = threadIdx.x & 7;
  const _Float16* aptr = A + (size_t)(m0 + a_row) * lda + a_seg * 16;
  const _Float16* bptr = B + (size_t)b_krow * ldb + n0 + b_seg * 16;

  // ---- prologue: stage K-tile 0 into buffer 0 ----
  async_copy_b256(aptr, &As[0][a_row][a_seg * 16]);
  {
    _Float16 btmp[16];
#pragma unroll
    for (int i = 0; i < 16; ++i) btmp[i] = bptr[i];
#pragma unroll
    for (int i = 0; i < 16; ++i) Bs[0][b_seg * 16 + i][b_krow] = btmp[i];
  }
  wait_async_all();
  __syncthreads();

  const int niter = K >> 5;
  for (int it = 0; it < niter; ++it) {
    const int cur = it & 1, nxt = cur ^ 1;
    const bool has_next = (it + 1) < niter;

    _Float16 btmp[16];
    if (has_next) {
      // kick off next A tile (async -> LDS) and next B tile (-> registers);
      // both overlap the WMMA work below.
      async_copy_b256(aptr + (size_t)(it + 1) * 32,
                      &As[nxt][a_row][a_seg * 16]);
      const _Float16* bsrc = bptr + (size_t)(it + 1) * 32 * ldb;
#pragma unroll
      for (int i = 0; i < 16; ++i) btmp[i] = bsrc[i];
    }

    // ---- compute on buffer `cur` ----
    v16h af[2], bf[4];
#pragma unroll
    for (int mi = 0; mi < 2; ++mi) {
      const _Float16* p = &As[cur][wm + mi * 16 + arow][akoff];
#pragma unroll
      for (int t = 0; t < 8; ++t) af[mi][t] = p[t];
#pragma unroll
      for (int t = 0; t < 8; ++t) af[mi][8 + t] = p[8 + t];
    }
#pragma unroll
    for (int ni = 0; ni < 4; ++ni) {
      const _Float16* p = &Bs[cur][wn + ni * 16 + bcol][bkoff];
#pragma unroll
      for (int t = 0; t < 16; ++t) bf[ni][t] = p[t];
    }
#pragma unroll
    for (int mi = 0; mi < 2; ++mi)
#pragma unroll
      for (int ni = 0; ni < 4; ++ni)
        acc[mi][ni] = wmma_f16(af[mi], bf[ni], acc[mi][ni]);

    if (has_next) {
#pragma unroll
      for (int i = 0; i < 16; ++i) Bs[nxt][b_seg * 16 + i][b_krow] = btmp[i];
    }
    wait_async_all();
    __syncthreads();
  }

  // Epilogue: C layout -> row = r + 8*(lane>=16), col = lane&15
  const int rhalf = (lane >> 4) << 3;
#pragma unroll
  for (int mi = 0; mi < 2; ++mi)
#pragma unroll
    for (int ni = 0; ni < 4; ++ni)
#pragma unroll
      for (int r = 0; r < 8; ++r) {
        const int row = m0 + wm + mi * 16 + r + rhalf;
        const int col = n0 + wn + ni * 16 + (lane & 15);
        const float v = acc[mi][ni][r] + bias[col];
        if constexpr (OUT_F32)
          Cf[(size_t)row * ldc + col] = v;
        else
          Ch[(size_t)row * ldc + col] = (_Float16)v;
      }
}

// ---------------------------------------------------------------------------
// Flash-style cross attention, one head per blockIdx.y, batch per blockIdx.z.
// 8 waves / block, each wave owns a 16-row Q tile (block covers 128 rows).
// S = Q*K^T via WMMA (dh=64 -> 2 k-steps), online softmax with -1e30 masking,
// P round-trips through per-wave LDS (C-layout -> A-layout), O += P*V via
// WMMA with V staged transposed in LDS.
// ---------------------------------------------------------------------------
__global__ void __launch_bounds__(256) attn_kernel(
    const _Float16* __restrict__ Qb, int ldq,
    const _Float16* __restrict__ Kb, const _Float16* __restrict__ Vb, int ldkv,
    const unsigned char* __restrict__ qmask,
    const unsigned char* __restrict__ kmask,
    _Float16* __restrict__ outb, int NQ, int NK, float scale) {
  __shared__ _Float16 Vst[64][36];       // transposed V tile [dh][key]
  __shared__ _Float16 Pst[8][16][36];    // per-wave P staging [row][col<32]

  const int b = blockIdx.z, h = blockIdx.y;
  const int lane = threadIdx.x & 31;
  const int w = threadIdx.x >> 5;
  const int q0 = blockIdx.x * 128 + w * 16;

  const _Float16* Q = Qb + ((size_t)b * NQ) * ldq + h * 64;
  const _Float16* Kp = Kb + ((size_t)b * NK) * ldkv + h * 64;
  const _Float16* Vp = Vb + ((size_t)b * NK) * ldkv + h * 64;

  const int arow  = lane & 15;
  const int akoff = (lane >> 4) * 16;
  const int rhalf = (lane >> 4) << 3;

  // Q fragments for dh=64 (2 k-steps of 32), held in registers
  v16h qf[2];
#pragma unroll
  for (int ks = 0; ks < 2; ++ks) {
    const _Float16* p = Q + (size_t)(q0 + arow) * ldq + ks * 32 + akoff;
#pragma unroll
    for (int t = 0; t < 8; ++t) qf[ks][t] = p[t];
#pragma unroll
    for (int t = 0; t < 8; ++t) qf[ks][8 + t] = p[8 + t];
  }

  bool qvalid[8];
#pragma unroll
  for (int r = 0; r < 8; ++r)
    qvalid[r] = qmask[(size_t)b * NQ + q0 + r + rhalf] != 0;

  float mrow[8], srow[8];
  v8f o[4];
#pragma unroll
  for (int r = 0; r < 8; ++r) { mrow[r] = -1e30f; srow[r] = 0.0f; }
#pragma unroll
  for (int nt = 0; nt < 4; ++nt) o[nt] = v8f_zero();

  for (int j = 0; j < NK; j += 32) {
    {  // cooperative transposed stage of V[j..j+31][0..63]
      const int key = threadIdx.x >> 3, seg = threadIdx.x & 7;
      const _Float16* vsrc = Vp + (size_t)(j + key) * ldkv + seg * 8;
#pragma unroll
      for (int i = 0; i < 8; ++i) Vst[seg * 8 + i][key] = vsrc[i];
    }
    __syncthreads();

    // S tiles for 32 keys (two 16-key sub-tiles)
    v8f st[2];
    bool km[2];
#pragma unroll
    for (int sub = 0; sub < 2; ++sub) {
      const int key = j + sub * 16 + (lane & 15);
      km[sub] = kmask[(size_t)b * NK + key] != 0;
      v8f s = v8f_zero();
#pragma unroll
      for (int ks = 0; ks < 2; ++ks) {
        // B = K^T: lane = key column, K-dim contiguous in memory
        v16h bf;
        const _Float16* p =
            Kp + (size_t)key * ldkv + ks * 32 + ((lane >> 4) * 16);
#pragma unroll
        for (int t = 0; t < 16; ++t) bf[t] = p[t];
        s = wmma_f16(qf[ks], bf, s);
      }
      st[sub] = s;
    }

    // Online softmax update (per row = vgpr r + 8*(lane>=16))
#pragma unroll
    for (int r = 0; r < 8; ++r) {
      const bool qv = qvalid[r];
      float s0 = (qv && km[0]) ? st[0][r] * scale : -1e30f;
      float s1 = (qv && km[1]) ? st[1][r] * scale : -1e30f;
      float mloc = fmaxf(s0, s1);
#pragma unroll
      for (int off = 1; off < 16; off <<= 1)
        mloc = fmaxf(mloc, __shfl_xor(mloc, off, 32));
      const float mnew = fmaxf(mrow[r], mloc);
      const float alpha = __expf(mrow[r] - mnew);
      const float p0 = __expf(s0 - mnew);
      const float p1 = __expf(s1 - mnew);
      float ps = p0 + p1;
#pragma unroll
      for (int off = 1; off < 16; off <<= 1) ps += __shfl_xor(ps, off, 32);
      srow[r] = srow[r] * alpha + ps;
      mrow[r] = mnew;
#pragma unroll
      for (int nt = 0; nt < 4; ++nt) o[nt][r] *= alpha;
      const int prow = r + rhalf;
      Pst[w][prow][lane & 15]        = (_Float16)p0;
      Pst[w][prow][16 + (lane & 15)] = (_Float16)p1;
    }
    // same-wave LDS RAW: DScnt keeps these in order (compiler inserts waits)

    // O += P(16x32) * V(32x64)
    v16h pf;
    {
      const _Float16* pp = &Pst[w][arow][akoff];
#pragma unroll
      for (int t = 0; t < 8; ++t) pf[t] = pp[t];
#pragma unroll
      for (int t = 0; t < 8; ++t) pf[8 + t] = pp[8 + t];
    }
#pragma unroll
    for (int nt = 0; nt < 4; ++nt) {
      v16h vf;
      const _Float16* vp2 = &Vst[nt * 16 + (lane & 15)][(lane >> 4) * 16];
#pragma unroll
      for (int t = 0; t < 16; ++t) vf[t] = vp2[t];
      o[nt] = wmma_f16(pf, vf, o[nt]);
    }
    __syncthreads();
  }

  // Epilogue: normalize; fully-masked rows (max still -1e30) -> 0
#pragma unroll
  for (int r = 0; r < 8; ++r) {
    const float inv = (mrow[r] > -1e29f) ? (1.0f / srow[r]) : 0.0f;
    const int qrow = q0 + r + rhalf;
    _Float16* op = outb + ((size_t)b * NQ + qrow) * 768 + h * 64;
#pragma unroll
    for (int nt = 0; nt < 4; ++nt)
      op[nt * 16 + (lane & 15)] = (_Float16)(o[nt][r] * inv);
  }
}

// ---------------------------------------------------------------------------
// Host launch
// ---------------------------------------------------------------------------
extern "C" void kernel_launch(void* const* d_in, const int* in_sizes, int n_in,
                              void* d_out, int out_size, void* d_ws,
                              size_t ws_size, hipStream_t stream) {
  (void)in_sizes; (void)n_in; (void)out_size; (void)ws_size;

  const float* protein = (const float*)d_in[0];
  const float* ligand  = (const float*)d_in[1];
  const unsigned char* pmask = (const unsigned char*)d_in[2];  // numpy bool
  const unsigned char* lmask = (const unsigned char*)d_in[3];
  const float* ln_p_scale = (const float*)d_in[4];
  const float* ln_p_bias  = (const float*)d_in[5];
  const float* ln_l_scale = (const float*)d_in[6];
  const float* ln_l_bias  = (const float*)d_in[7];
  const float* W_pqkv = (const float*)d_in[8];
  const float* b_pqkv = (const float*)d_in[9];
  const float* W_lqkv = (const float*)d_in[10];
  const float* b_lqkv = (const float*)d_in[11];
  const float* W_pout = (const float*)d_in[12];
  const float* b_pout = (const float*)d_in[13];
  const float* W_lout = (const float*)d_in[14];
  const float* b_lout = (const float*)d_in[15];

  constexpr int Bb = 16, P = 2048, L = 256, D = 768, H = 12;
  constexpr size_t MP = (size_t)Bb * P;  // 32768 protein rows
  constexpr size_t ML = (size_t)Bb * L;  // 4096 ligand rows

  _Float16* ws = (_Float16*)d_ws;
  size_t off = 0;
  _Float16* pnorm = ws + off; off += MP * D;
  _Float16* lnorm = ws + off; off += ML * D;
  _Float16* wpqkv = ws + off; off += (size_t)D * 3 * D;
  _Float16* wlqkv = ws + off; off += (size_t)D * 3 * D;
  _Float16* wpout = ws + off; off += (size_t)D * D;
  _Float16* wlout = ws + off; off += (size_t)D * D;
  _Float16* pqkv  = ws + off; off += MP * 3 * D;
  _Float16* lqkv  = ws + off; off += ML * 3 * D;
  _Float16* pattn = ws + off; off += MP * D;
  _Float16* lattn = ws + off; off += ML * D;

  // 1. LayerNorm -> f16
  ln_f16_kernel<<<dim3((unsigned)MP), 256, 0, stream>>>(protein, ln_p_scale,
                                                        ln_p_bias, pnorm);
  ln_f16_kernel<<<dim3((unsigned)ML), 256, 0, stream>>>(ligand, ln_l_scale,
                                                        ln_l_bias, lnorm);

  // 2. Weight conversion -> f16
  f32_to_f16_kernel<<<(D * 3 * D + 255) / 256, 256, 0, stream>>>(W_pqkv, wpqkv, D * 3 * D);
  f32_to_f16_kernel<<<(D * 3 * D + 255) / 256, 256, 0, stream>>>(W_lqkv, wlqkv, D * 3 * D);
  f32_to_f16_kernel<<<(D * D + 255) / 256, 256, 0, stream>>>(W_pout, wpout, D * D);
  f32_to_f16_kernel<<<(D * D + 255) / 256, 256, 0, stream>>>(W_lout, wlout, D * D);

  // 3. QKV projections (f16 out)
  gemm_bias_kernel<false><<<dim3(3 * D / 128, (unsigned)(MP / 128)), 256, 0, stream>>>(
      pnorm, D, wpqkv, 3 * D, b_pqkv, nullptr, pqkv, 3 * D, D);
  gemm_bias_kernel<false><<<dim3(3 * D / 128, (unsigned)(ML / 128)), 256, 0, stream>>>(
      lnorm, D, wlqkv, 3 * D, b_lqkv, nullptr, lqkv, 3 * D, D);

  // 4. Cross attention (qkv buffers: [row, 3*768] with Q|K|V sections)
  const float scale = 0.125f;  // 64^-0.5
  attn_kernel<<<dim3(P / 128, H, Bb), 256, 0, stream>>>(
      pqkv, 3 * D, lqkv + D, lqkv + 2 * D, 3 * D, pmask, lmask, pattn, P, L, scale);
  attn_kernel<<<dim3(L / 128, H, Bb), 256, 0, stream>>>(
      lqkv, 3 * D, pqkv + D, pqkv + 2 * D, 3 * D, lmask, pmask, lattn, L, P, scale);

  // 5. Output projections (f32 out, directly into d_out)
  float* pout = (float*)d_out;
  float* lout = (float*)d_out + MP * D;
  gemm_bias_kernel<true><<<dim3(D / 128, (unsigned)(MP / 128)), 256, 0, stream>>>(
      pattn, D, wpout, D, b_pout, pout, nullptr, D, D);
  gemm_bias_kernel<true><<<dim3(D / 128, (unsigned)(ML / 128)), 256, 0, stream>>>(
      lattn, D, wlout, D, b_lout, lout, nullptr, D, D);
}
